// ContrastiveModel_56392920596638
// MI455X (gfx1250) — compile-verified, matrix operand order checked by reading, not run on hardware
//
#include <hip/hip_runtime.h>
#include <hip/hip_bf16.h>

// ---------------------------------------------------------------------------
// Model constants (from the reference)
// ---------------------------------------------------------------------------
constexpr int B_  = 16;
constexpr int D_  = 512;
constexpr int H_  = 8;
constexpr int FF_ = 2048;
constexpr int NC_ = 4;
constexpr int NF_ = 30;
constexpr int NN_ = 34;
constexpr int FL_ = 64;
constexpr int HL_ = 32;
constexpr int NLAYERS_ = 2;

typedef __attribute__((ext_vector_type(16))) _Float16 v16h;
typedef __attribute__((ext_vector_type(8)))  _Float16 v8h;
typedef __attribute__((ext_vector_type(8)))  float    v8f;

union AFrag16 { v16h v; v8h h[2]; };

__device__ __forceinline__ v8f zero_v8f() {
  v8f z;
#pragma unroll
  for (int e = 0; e < 8; ++e) z[e] = 0.0f;
  return z;
}

// ---------------------------------------------------------------------------
// f32 -> f16 conversion (weights)
// ---------------------------------------------------------------------------
__global__ void cvt_f16_kernel(const float* __restrict__ src,
                               _Float16* __restrict__ dst, int n) {
  int i = blockIdx.x * 256 + threadIdx.x;
  if (i < n) dst[i] = (_Float16)src[i];
}

// ---------------------------------------------------------------------------
// Embedding gather + sqrt(D) scale + sinusoidal positional encoding
// ---------------------------------------------------------------------------
__global__ __launch_bounds__(256)
void embed_kernel(const int* __restrict__ ids, const float* __restrict__ emb,
                  float* __restrict__ x32, _Float16* __restrict__ x16, int L) {
  const int row = blockIdx.x;
  const int id  = ids[row];
  const int pos = row % L;
  const size_t base = (size_t)row * D_;
  for (int d = threadIdx.x; d < D_; d += 256) {
    int pair = d >> 1;
    float ang = (float)pos * __powf(10000.0f, -(float)(2 * pair) / (float)D_);
    float pe  = (d & 1) ? __cosf(ang) : __sinf(ang);
    float v   = emb[(size_t)id * D_ + d] * 22.62741699796952f + pe;
    x32[base + d] = v;
    x16[base + d] = (_Float16)v;
  }
}

// ---------------------------------------------------------------------------
// Tiled WMMA GEMM: out[M,N] = A[M,K](f16) * W[K,N](f16) + bias[N]
// EPI: 0 = f16 out, 1 = f16 + ReLU out, 2 = f32 out
// Block tile 128x128, BK=32, 8 waves (4Mx2N), each wave 2x4 16x16 tiles.
// ---------------------------------------------------------------------------
#define LDS_STRIDE 40

template <int EPI>
__global__ __launch_bounds__(256)
void gemm_wmma_kernel(const _Float16* __restrict__ A,
                      const _Float16* __restrict__ W,
                      const float* __restrict__ bias,
                      void* __restrict__ out, int M, int N, int K) {
  __shared__ alignas(16) _Float16 lA[2][128][LDS_STRIDE];
  __shared__ alignas(16) _Float16 lB[2][128][LDS_STRIDE];  // lB[n][k] = W^T tile

  const int tid   = threadIdx.x;
  const int lane  = tid & 31;
  const int wid   = tid >> 5;
  const int waveM = wid & 3;   // 0..3
  const int waveN = wid >> 2;  // 0..1
  const int bm    = blockIdx.x * 128;
  const int bn    = blockIdx.y * 128;
  const int khb   = (lane >> 4) * 8;  // ISA 16-bit A/B layout: K chunk base per lane half

  v8f acc[2][4];
#pragma unroll
  for (int i = 0; i < 2; ++i)
#pragma unroll
    for (int j = 0; j < 4; ++j) acc[i][j] = zero_v8f();

  auto stageA = [&](int buf, int kt) {
#pragma unroll
    for (int t = 0; t < 2; ++t) {
      int a   = tid + t * 256;   // 512 chunk loads of 8 halves
      int row = a >> 2;          // 128 rows
      int ck  = (a & 3) * 8;     // 4 chunks of 8 per 32-wide row
      const v8h* src = (const v8h*)(A + (size_t)(bm + row) * K + kt + ck);
      *(v8h*)&lA[buf][row][ck] = *src;
    }
    if (kt + 64 < K)
      __builtin_prefetch(A + (size_t)(bm + (tid >> 1)) * K + kt + 64, 0, 1);
  };
  auto stageB = [&](int buf, int kt) {
#pragma unroll
    for (int t = 0; t < 2; ++t) {
      int a  = tid + t * 256;    // 512 chunk loads of 8 halves
      int k  = a >> 4;           // 32 k rows
      int nc = (a & 15) * 8;     // 16 chunks of 8 n
      v8h vals = *(const v8h*)(W + (size_t)(kt + k) * N + bn + nc);
#pragma unroll
      for (int e = 0; e < 8; ++e) lB[buf][nc + e][k] = vals[e];  // transpose into LDS
    }
  };
  auto compute = [&](int buf) {
    AFrag16 afr[2], bfr[4];
#pragma unroll
    for (int i = 0; i < 2; ++i) {
      int m = waveM * 32 + i * 16 + (lane & 15);
      afr[i].h[0] = *(const v8h*)&lA[buf][m][khb];
      afr[i].h[1] = *(const v8h*)&lA[buf][m][khb + 16];
    }
#pragma unroll
    for (int j = 0; j < 4; ++j) {
      int n = waveN * 64 + j * 16 + (lane & 15);
      bfr[j].h[0] = *(const v8h*)&lB[buf][n][khb];
      bfr[j].h[1] = *(const v8h*)&lB[buf][n][khb + 16];
    }
#pragma unroll
    for (int i = 0; i < 2; ++i)
#pragma unroll
      for (int j = 0; j < 4; ++j)
        acc[i][j] = __builtin_amdgcn_wmma_f32_16x16x32_f16(
            false, afr[i].v, false, bfr[j].v, (short)0, acc[i][j], false, false);
  };

  int buf = 0;
  stageA(0, 0);
  stageB(0, 0);
  for (int kt = 0; kt < K; kt += 32) {
    __syncthreads();
    if (kt + 32 < K) { stageA(buf ^ 1, kt + 32); stageB(buf ^ 1, kt + 32); }
    compute(buf);
    buf ^= 1;
  }

#pragma unroll
  for (int i = 0; i < 2; ++i)
#pragma unroll
    for (int j = 0; j < 4; ++j)
#pragma unroll
      for (int v = 0; v < 8; ++v) {
        int row = bm + waveM * 32 + i * 16 + (lane >> 4) * 8 + v;
        int col = bn + waveN * 64 + j * 16 + (lane & 15);
        float val = acc[i][j][v] + bias[col];
        if (EPI == 1) val = fmaxf(val, 0.0f);
        if (EPI == 2) ((float*)out)[(size_t)row * N + col] = val;
        else          ((_Float16*)out)[(size_t)row * N + col] = (_Float16)val;
      }
}

// ---------------------------------------------------------------------------
// Fused attention: per (sequence, head). qkv rows are [Q(512) K(512) V(512)].
// scores = softmax(Q K^T / 8), out = scores @ V. L in {64, 32}. 4 waves.
// ---------------------------------------------------------------------------
__global__ __launch_bounds__(128)
void attention_kernel(const _Float16* __restrict__ qkv,
                      _Float16* __restrict__ out, int L) {
  __shared__ alignas(16) _Float16 sQ[64][72];   // reused for P after softmax
  __shared__ alignas(16) _Float16 sK[64][72];
  __shared__ alignas(16) _Float16 sVt[64][72];  // sVt[d][key]
  __shared__ float sS[64][65];

  const int seq  = blockIdx.x;
  const int h    = blockIdx.y;
  const int tid  = threadIdx.x;
  const int lane = tid & 31;
  const int wv   = tid >> 5;
  const int khb  = (lane >> 4) * 8;
  const size_t base = (size_t)seq * L * (3 * D_);

  for (int a = tid; a < L * 8; a += 128) {
    int r = a >> 3, c = (a & 7) * 8;
    *(v8h*)&sQ[r][c] = *(const v8h*)(qkv + base + (size_t)r * (3 * D_) + h * 64 + c);
    *(v8h*)&sK[r][c] = *(const v8h*)(qkv + base + (size_t)r * (3 * D_) + D_ + h * 64 + c);
    v8h vv = *(const v8h*)(qkv + base + (size_t)r * (3 * D_) + 2 * D_ + h * 64 + c);
#pragma unroll
    for (int e = 0; e < 8; ++e) sVt[c + e][r] = vv[e];
  }
  __syncthreads();

  // S = Q K^T * (1/8): wave wv owns query rows [wv*16, wv*16+16)
  if (wv * 16 < L) {
    for (int nt = 0; nt < L / 16; ++nt) {
      v8f acc = zero_v8f();
#pragma unroll
      for (int kk = 0; kk < 64; kk += 32) {
        AFrag16 af, bf;
        int m = wv * 16 + (lane & 15);
        af.h[0] = *(const v8h*)&sQ[m][kk + khb];
        af.h[1] = *(const v8h*)&sQ[m][kk + khb + 16];
        int n = nt * 16 + (lane & 15);
        bf.h[0] = *(const v8h*)&sK[n][kk + khb];
        bf.h[1] = *(const v8h*)&sK[n][kk + khb + 16];
        acc = __builtin_amdgcn_wmma_f32_16x16x32_f16(
            false, af.v, false, bf.v, (short)0, acc, false, false);
      }
#pragma unroll
      for (int v = 0; v < 8; ++v)
        sS[wv * 16 + (lane >> 4) * 8 + v][nt * 16 + (lane & 15)] = acc[v] * 0.125f;
    }
  }
  __syncthreads();

  // row softmax -> P (f16, stored back into sQ)
  if (tid < L) {
    float mx = -1e30f;
    for (int j = 0; j < L; ++j) mx = fmaxf(mx, sS[tid][j]);
    float sum = 0.0f;
    for (int j = 0; j < L; ++j) { float e = __expf(sS[tid][j] - mx); sS[tid][j] = e; sum += e; }
    float inv = 1.0f / sum;
    for (int j = 0; j < L; ++j) sQ[tid][j] = (_Float16)(sS[tid][j] * inv);
  }
  __syncthreads();

  // O = P @ V
  if (wv * 16 < L) {
    for (int nt = 0; nt < 4; ++nt) {
      v8f acc = zero_v8f();
      for (int kk = 0; kk < L; kk += 32) {
        AFrag16 af, bf;
        int m = wv * 16 + (lane & 15);
        af.h[0] = *(const v8h*)&sQ[m][kk + khb];
        af.h[1] = *(const v8h*)&sQ[m][kk + khb + 16];
        int n = nt * 16 + (lane & 15);
        bf.h[0] = *(const v8h*)&sVt[n][kk + khb];
        bf.h[1] = *(const v8h*)&sVt[n][kk + khb + 16];
        acc = __builtin_amdgcn_wmma_f32_16x16x32_f16(
            false, af.v, false, bf.v, (short)0, acc, false, false);
      }
#pragma unroll
      for (int v = 0; v < 8; ++v) {
        int m = wv * 16 + (lane >> 4) * 8 + v;
        out[((size_t)seq * L + m) * D_ + h * 64 + nt * 16 + (lane & 15)] = (_Float16)acc[v];
      }
    }
  }
}

// ---------------------------------------------------------------------------
// x = LayerNorm(x + y) * g + b ; also emit f16 copy. One block per row (D=512).
// ---------------------------------------------------------------------------
__global__ __launch_bounds__(256)
void residual_ln_kernel(float* __restrict__ x, const float* __restrict__ y,
                        const float* __restrict__ g, const float* __restrict__ b,
                        _Float16* __restrict__ x16) {
  const int tid = threadIdx.x;
  const size_t base = (size_t)blockIdx.x * D_;
  __shared__ float red[256];
  float v0 = x[base + tid] + y[base + tid];
  float v1 = x[base + 256 + tid] + y[base + 256 + tid];
  red[tid] = v0 + v1;
  __syncthreads();
  for (int st = 128; st > 0; st >>= 1) { if (tid < st) red[tid] += red[tid + st]; __syncthreads(); }
  float mean = red[0] * (1.0f / D_);
  __syncthreads();
  float d0 = v0 - mean, d1 = v1 - mean;
  red[tid] = d0 * d0 + d1 * d1;
  __syncthreads();
  for (int st = 128; st > 0; st >>= 1) { if (tid < st) red[tid] += red[tid + st]; __syncthreads(); }
  float rstd = rsqrtf(red[0] * (1.0f / D_) + 1e-5f);
  float o0 = d0 * rstd * g[tid] + b[tid];
  float o1 = d1 * rstd * g[256 + tid] + b[256 + tid];
  x[base + tid] = o0;       x[base + 256 + tid] = o1;
  x16[base + tid] = (_Float16)o0; x16[base + 256 + tid] = (_Float16)o1;
}

// ---------------------------------------------------------------------------
// Masked mean over sequence: one block per (batch, seq-slot) row group.
// ---------------------------------------------------------------------------
__global__ __launch_bounds__(256)
void masked_mean_kernel(const float* __restrict__ x, const float* __restrict__ mask,
                        float* __restrict__ seq_out, int L) {
  const int bn = blockIdx.x;
  float cnt = 0.0f;
  for (int l = 0; l < L; ++l) cnt += mask[bn * L + l];
  cnt = fmaxf(cnt, 1e-9f);
  const float inv = 1.0f / cnt;
  for (int d = threadIdx.x; d < D_; d += 256) {
    float s = 0.0f;
    for (int l = 0; l < L; ++l)
      s += x[((size_t)bn * L + l) * D_ + d] * mask[bn * L + l];
    seq_out[(size_t)bn * D_ + d] = s * inv;
  }
}

// ---------------------------------------------------------------------------
// Row dot: out[r] = rows[r] . w  (D=512). One block per row.
// ---------------------------------------------------------------------------
__global__ __launch_bounds__(256)
void rowdot_kernel(const float* __restrict__ rows, const float* __restrict__ w,
                   float* __restrict__ out) {
  const int r = blockIdx.x, tid = threadIdx.x;
  __shared__ float red[256];
  const size_t base = (size_t)r * D_;
  red[tid] = rows[base + tid] * w[tid] + rows[base + 256 + tid] * w[256 + tid];
  __syncthreads();
  for (int st = 128; st > 0; st >>= 1) { if (tid < st) red[tid] += red[tid + st]; __syncthreads(); }
  if (tid == 0) out[r] = red[0];
}

// ---------------------------------------------------------------------------
// aa[b,i,j] = fA1[b,j] + fA2[b,i] + sum_d |f_j - f_i| A3 + sum_d f_i A4 f_j + bias
// ---------------------------------------------------------------------------
__global__ __launch_bounds__(64)
void aa_pair_kernel(const float* __restrict__ fseq, const float* __restrict__ fA1,
                    const float* __restrict__ fA2, const float* __restrict__ A3,
                    const float* __restrict__ A4, const float* __restrict__ bias,
                    float* __restrict__ aa) {
  const int idx = blockIdx.x;
  const int b = idx / (NF_ * NF_);
  const int rem = idx % (NF_ * NF_);
  const int i = rem / NF_, j = rem % NF_;
  const float* fi = fseq + ((size_t)b * NF_ + i) * D_;
  const float* fj = fseq + ((size_t)b * NF_ + j) * D_;
  const int tid = threadIdx.x;
  float t = 0.0f;
  for (int d = tid; d < D_; d += 64)
    t += fabsf(fj[d] - fi[d]) * A3[d] + fi[d] * A4[d] * fj[d];
  __shared__ float red[64];
  red[tid] = t;
  __syncthreads();
  for (int st = 32; st > 0; st >>= 1) { if (tid < st) red[tid] += red[tid + st]; __syncthreads(); }
  if (tid == 0) aa[idx] = fA1[b * NF_ + j] + fA2[b * NF_ + i] + red[0] + bias[0];
}

// softmax over flattened NF*NF per batch, then zero diagonal
__global__ __launch_bounds__(256)
void aa_softmax_kernel(const float* __restrict__ aa, float* __restrict__ aa_sm) {
  const int b = blockIdx.x, tid = threadIdx.x;
  const float* src = aa + (size_t)b * NF_ * NF_;
  __shared__ float red[256];
  float mx = -1e30f;
  for (int e = tid; e < NF_ * NF_; e += 256) mx = fmaxf(mx, src[e]);
  red[tid] = mx;
  __syncthreads();
  for (int st = 128; st > 0; st >>= 1) { if (tid < st) red[tid] = fmaxf(red[tid], red[tid + st]); __syncthreads(); }
  mx = red[0];
  __syncthreads();
  float s = 0.0f;
  for (int e = tid; e < NF_ * NF_; e += 256) s += __expf(src[e] - mx);
  red[tid] = s;
  __syncthreads();
  for (int st = 128; st > 0; st >>= 1) { if (tid < st) red[tid] += red[tid + st]; __syncthreads(); }
  const float inv = 1.0f / red[0];
  for (int e = tid; e < NF_ * NF_; e += 256) {
    int i = e / NF_, j = e % NF_;
    float v = __expf(src[e] - mx) * inv;
    aa_sm[(size_t)b * NF_ * NF_ + e] = (i == j) ? 0.0f : v;
  }
}

// ---------------------------------------------------------------------------
// qa[b,c,k] = fS1[b,k] + hS2[b,c] + sum_d |f_k - h_c| S3 + sum_d f_k S4 h_c + sb
// ---------------------------------------------------------------------------
__global__ __launch_bounds__(64)
void qa_pair_kernel(const float* __restrict__ fseq, const float* __restrict__ hseq,
                    const float* __restrict__ fS1, const float* __restrict__ hS2,
                    const float* __restrict__ S3, const float* __restrict__ S4,
                    const float* __restrict__ sb, float* __restrict__ qa) {
  const int idx = blockIdx.x;
  const int b = idx / (NC_ * NF_);
  const int rem = idx % (NC_ * NF_);
  const int c = rem / NF_, k = rem % NF_;
  const float* fk = fseq + ((size_t)b * NF_ + k) * D_;
  const float* hc = hseq + ((size_t)b * NC_ + c) * D_;
  const int tid = threadIdx.x;
  float t = 0.0f;
  for (int d = tid; d < D_; d += 64)
    t += fabsf(fk[d] - hc[d]) * S3[d] + fk[d] * S4[d] * hc[d];
  __shared__ float red[64];
  red[tid] = t;
  __syncthreads();
  for (int st = 32; st > 0; st >>= 1) { if (tid < st) red[tid] += red[tid + st]; __syncthreads(); }
  if (tid == 0) qa[idx] = fS1[b * NF_ + k] + hS2[b * NC_ + c] + red[0] + sb[0];
}

// softmax over k (NF) for each of the B*NC rows; 64 threads, one row each
__global__ void qa_softmax_kernel(const float* __restrict__ qa, float* __restrict__ sim) {
  const int r = threadIdx.x;  // B*NC == 64
  const float* row = qa + (size_t)r * NF_;
  float mx = -1e30f;
  for (int k = 0; k < NF_; ++k) mx = fmaxf(mx, row[k]);
  float s = 0.0f;
  for (int k = 0; k < NF_; ++k) s += __expf(row[k] - mx);
  const float inv = 1.0f / s;
  float* o = sim + (size_t)r * NF_;
  for (int k = 0; k < NF_; ++k) o[k] = __expf(row[k] - mx) * inv;
}

// ---------------------------------------------------------------------------
// Edge weights: W = -p0*aa_ovl + p1*aa_sim' + p2*qa_ovl + p3*qa_sim' + qq
// ---------------------------------------------------------------------------
__global__ void edge_weight_kernel(const float* __restrict__ aa_ovl,
                                   const float* __restrict__ aa_sim_in,
                                   const float* __restrict__ qa_ovl,
                                   const float* __restrict__ qq,
                                   const float* __restrict__ aa_sm,
                                   const float* __restrict__ sim,
                                   const float* __restrict__ p_aa_ov,
                                   const float* __restrict__ p_aa_sim,
                                   const float* __restrict__ p_qa_ov,
                                   const float* __restrict__ p_qa_sim,
                                   float* __restrict__ Wout) {
  const int idx = blockIdx.x * 256 + threadIdx.x;
  if (idx >= B_ * NN_ * NN_) return;
  const int b = idx / (NN_ * NN_);
  const int rem = idx % (NN_ * NN_);
  const int u = rem / NN_, v = rem % NN_;
  float aasim = (u >= NC_ && v >= NC_)
      ? aa_sm[((size_t)b * NF_ + (u - NC_)) * NF_ + (v - NC_)]
      : aa_sim_in[idx];
  float qasim = 0.0f;
  if (u >= NC_ && v < NC_)      qasim = sim[((size_t)b * NC_ + v) * NF_ + (u - NC_)];
  else if (u < NC_ && v >= NC_) qasim = sim[((size_t)b * NC_ + u) * NF_ + (v - NC_)];
  Wout[idx] = -p_aa_ov[0] * aa_ovl[idx] + p_aa_sim[0] * aasim
            +  p_qa_ov[0] * qa_ovl[idx] + p_qa_sim[0] * qasim + qq[idx];
}

// ---------------------------------------------------------------------------
// Projected-gradient edge solver: 100 iterations entirely in LDS, 1 block / b.
// ---------------------------------------------------------------------------
__global__ __launch_bounds__(256)
void solve_edges_kernel(const float* __restrict__ Wg, float* __restrict__ final_pred) {
  const int b = blockIdx.x, tid = threadIdx.x;
  __shared__ float Wl[NN_ * NN_], adj[NN_ * NN_], E[NN_ * NN_], En[NN_ * NN_];
  __shared__ float red[256], red2[256];
  for (int e = tid; e < NN_ * NN_; e += 256) {
    float w = Wg[(size_t)b * NN_ * NN_ + e];
    Wl[e] = w;
    adj[e] = (w != 0.0f) ? 1.0f : 0.0f;
    E[e] = 0.5f;
  }
  __syncthreads();
  const float lr = 0.05f, cc = 10.0f;
  for (int it = 0; it < 100; ++it) {
    float s = 0.0f;
    for (int e = tid; e < NN_ * NN_; e += 256) s += adj[e] * E[e];
    float d = (tid < NC_) ? E[tid * NN_ + tid] : 0.0f;
    red[tid] = s; red2[tid] = d;
    __syncthreads();
    for (int st = 128; st > 0; st >>= 1) {
      if (tid < st) { red[tid] += red[tid + st]; red2[tid] += red2[tid + st]; }
      __syncthreads();
    }
    const float pen_s = 2.0f * cc * fmaxf(red[0] - 6.0f, 0.0f);
    const float pen_d = 2.0f * cc * (red2[0] - 1.0f);
    __syncthreads();
    for (int e = tid; e < NN_ * NN_; e += 256) {
      int i = e / NN_, j = e % NN_;
      float diag4 = (i == j && i < NC_) ? 1.0f : 0.0f;
      En[e] = E[e] + lr * (Wl[e] - pen_s * adj[e] - pen_d * diag4);
    }
    __syncthreads();
    for (int e = tid; e < NN_ * NN_; e += 256) {
      int i = e / NN_, j = e % NN_;
      float v = 0.5f * (En[e] + En[j * NN_ + i]);
      E[e] = fminf(fmaxf(v, 0.0f), 1.0f);
    }
    __syncthreads();
  }
  if (tid < NC_) final_pred[b * NC_ + tid] = E[tid * NN_ + tid];
}

// ---------------------------------------------------------------------------
// Final loss: ce(log_softmax(final_pred), labels) + mse(sim, softmax(gold))
// out[0] = loss, out[1..64] = final_pred
// ---------------------------------------------------------------------------
__global__ __launch_bounds__(256)
void loss_kernel(const float* __restrict__ sim, const float* __restrict__ gold,
                 const int* __restrict__ labels, const float* __restrict__ final_pred,
                 float* __restrict__ out) {
  const int tid = threadIdx.x;
  __shared__ float red[256];
  float msep = 0.0f;
  if (tid < B_ * NC_) {
    const float* gr = gold + (size_t)tid * NF_;
    const float* sr = sim + (size_t)tid * NF_;
    float mx = -1e30f;
    for (int k = 0; k < NF_; ++k) mx = fmaxf(mx, gr[k]);
    float s = 0.0f;
    for (int k = 0; k < NF_; ++k) s += __expf(gr[k] - mx);
    float inv = 1.0f / s;
    for (int k = 0; k < NF_; ++k) {
      float g = __expf(gr[k] - mx) * inv;
      float dd = sr[k] - g;
      msep += dd * dd;
    }
  }
  red[tid] = msep;
  __syncthreads();
  for (int st = 128; st > 0; st >>= 1) { if (tid < st) red[tid] += red[tid + st]; __syncthreads(); }
  const float mse = red[0] / (float)(B_ * NC_ * NF_);
  __syncthreads();
  float cep = 0.0f;
  if (tid < B_) {
    const float* fp = final_pred + tid * NC_;
    float mx = -1e30f;
    for (int c = 0; c < NC_; ++c) mx = fmaxf(mx, fp[c]);
    float s = 0.0f;
    for (int c = 0; c < NC_; ++c) s += __expf(fp[c] - mx);
    float lse = mx + logf(s);
    cep = -(fp[labels[tid]] - lse);
  }
  red[tid] = cep;
  __syncthreads();
  for (int st = 128; st > 0; st >>= 1) { if (tid < st) red[tid] += red[tid + st]; __syncthreads(); }
  if (tid == 0) out[0] = red[0] / (float)B_ + mse;
  if (tid < B_ * NC_) out[1 + tid] = final_pred[tid];
}

// ---------------------------------------------------------------------------
// Host orchestration
// ---------------------------------------------------------------------------
extern "C" void kernel_launch(void* const* d_in, const int* in_sizes, int n_in,
                              void* d_out, int out_size, void* d_ws, size_t ws_size,
                              hipStream_t stream) {
  (void)in_sizes; (void)n_in; (void)out_size; (void)ws_size;

  const int*   hyp_ids   = (const int*)d_in[0];
  const float* hyp_mask  = (const float*)d_in[1];
  const float* fact_mask = (const float*)d_in[2];
  const int*   fact_ids  = (const int*)d_in[3];
  const float* aa_ovl    = (const float*)d_in[5];
  const float* aa_sim_in = (const float*)d_in[6];
  const float* qa_ovl    = (const float*)d_in[8];
  const float* qq        = (const float*)d_in[11];
  const int*   labels    = (const int*)d_in[12];
  const float* gold      = (const float*)d_in[13];
  const float* emb       = (const float*)d_in[14];
  const float* Wqkv      = (const float*)d_in[15];
  const float* bqkv      = (const float*)d_in[16];
  const float* Wo        = (const float*)d_in[17];
  const float* bo        = (const float*)d_in[18];
  const float* ln1_g     = (const float*)d_in[19];
  const float* ln1_b     = (const float*)d_in[20];
  const float* Wff1      = (const float*)d_in[21];
  const float* bff1      = (const float*)d_in[22];
  const float* Wff2      = (const float*)d_in[23];
  const float* bff2      = (const float*)d_in[24];
  const float* ln2_g     = (const float*)d_in[25];
  const float* ln2_b     = (const float*)d_in[26];
  const float* scoreW    = (const float*)d_in[27];
  const float* score_b   = (const float*)d_in[28];
  const float* absW      = (const float*)d_in[29];
  const float* abs_b     = (const float*)d_in[30];
  const float* p_aa_ov   = (const float*)d_in[31];
  const float* p_aa_sim  = (const float*)d_in[32];
  const float* p_qa_ov   = (const float*)d_in[33];
  const float* p_qa_sim  = (const float*)d_in[34];

  const int Mf = B_ * NF_ * FL_;   // 30720 fact token rows
  const int Mh = B_ * NC_ * HL_;   // 2048 hypothesis token rows
  const int maxM = Mf;

  // workspace carve-out
  char* ws = (char*)d_ws;
  size_t off = 0;
  auto take = [&](size_t bytes) -> char* {
    char* p = ws + off;
    off += (bytes + 255) & ~(size_t)255;
    return p;
  };
  float*     x32    = (float*)take((size_t)maxM * D_ * 4);
  _Float16*  x16    = (_Float16*)take((size_t)maxM * D_ * 2);
  _Float16*  big16  = (_Float16*)take((size_t)maxM * FF_ * 2);   // qkv / ff1 acts
  float*     tmp32  = (float*)take((size_t)maxM * D_ * 4);
  _Float16*  att16  = (_Float16*)take((size_t)maxM * D_ * 2);
  _Float16*  wqkv16 = (_Float16*)take((size_t)NLAYERS_ * D_ * 3 * D_ * 2);
  _Float16*  wo16   = (_Float16*)take((size_t)NLAYERS_ * D_ * D_ * 2);
  _Float16*  wf1_16 = (_Float16*)take((size_t)NLAYERS_ * D_ * FF_ * 2);
  _Float16*  wf2_16 = (_Float16*)take((size_t)NLAYERS_ * FF_ * D_ * 2);
  float*     fseq   = (float*)take((size_t)B_ * NF_ * D_ * 4);
  float*     hseq   = (float*)take((size_t)B_ * NC_ * D_ * 4);
  float*     fA1    = (float*)take((size_t)B_ * NF_ * 4);
  float*     fA2    = (float*)take((size_t)B_ * NF_ * 4);
  float*     fS1    = (float*)take((size_t)B_ * NF_ * 4);
  float*     hS2    = (float*)take((size_t)B_ * NC_ * 4);
  float*     aabuf  = (float*)take((size_t)B_ * NF_ * NF_ * 4);
  float*     aasm   = (float*)take((size_t)B_ * NF_ * NF_ * 4);
  float*     qabuf  = (float*)take((size_t)B_ * NC_ * NF_ * 4);
  float*     simbuf = (float*)take((size_t)B_ * NC_ * NF_ * 4);
  float*     wedge  = (float*)take((size_t)B_ * NN_ * NN_ * 4);
  float*     fpred  = (float*)take((size_t)B_ * NC_ * 4);

  // weight conversion f32 -> f16
  auto cvt = [&](const float* s, _Float16* d, int n) {
    cvt_f16_kernel<<<(n + 255) / 256, 256, 0, stream>>>(s, d, n);
  };
  cvt(Wqkv, wqkv16, NLAYERS_ * D_ * 3 * D_);
  cvt(Wo,   wo16,   NLAYERS_ * D_ * D_);
  cvt(Wff1, wf1_16, NLAYERS_ * D_ * FF_);
  cvt(Wff2, wf2_16, NLAYERS_ * FF_ * D_);

  // full encoder pass for one path
  auto encode = [&](const int* ids, const float* mask, int M, int L, int numSeq,
                    float* seq_out) {
    embed_kernel<<<M, 256, 0, stream>>>(ids, emb, x32, x16, L);
    for (int l = 0; l < NLAYERS_; ++l) {
      // qkv = x @ Wqkv + bqkv  (f16 out)
      gemm_wmma_kernel<0><<<dim3(M / 128, (3 * D_) / 128), 256, 0, stream>>>(
          x16, wqkv16 + (size_t)l * D_ * 3 * D_, bqkv + l * 3 * D_, big16, M, 3 * D_, D_);
      // attention per (seq, head)
      attention_kernel<<<dim3(numSeq, H_), 128, 0, stream>>>(big16, att16, L);
      // o = att @ Wo + bo  (f32 out)
      gemm_wmma_kernel<2><<<dim3(M / 128, D_ / 128), 256, 0, stream>>>(
          att16, wo16 + (size_t)l * D_ * D_, bo + l * D_, tmp32, M, D_, D_);
      residual_ln_kernel<<<M, 256, 0, stream>>>(x32, tmp32, ln1_g + l * D_, ln1_b + l * D_, x16);
      // f = relu(x @ Wff1 + bff1)  (f16 out)
      gemm_wmma_kernel<1><<<dim3(M / 128, FF_ / 128), 256, 0, stream>>>(
          x16, wf1_16 + (size_t)l * D_ * FF_, bff1 + l * FF_, big16, M, FF_, D_);
      // y = f @ Wff2 + bff2  (f32 out)
      gemm_wmma_kernel<2><<<dim3(M / 128, D_ / 128), 256, 0, stream>>>(
          big16, wf2_16 + (size_t)l * FF_ * D_, bff2 + l * D_, tmp32, M, D_, FF_);
      residual_ln_kernel<<<M, 256, 0, stream>>>(x32, tmp32, ln2_g + l * D_, ln2_b + l * D_, x16);
    }
    masked_mean_kernel<<<numSeq, 256, 0, stream>>>(x32, mask, seq_out, L);
  };

  encode(fact_ids, fact_mask, Mf, FL_, B_ * NF_, fseq);
  encode(hyp_ids,  hyp_mask,  Mh, HL_, B_ * NC_, hseq);

  // abstract-abstract scoring
  rowdot_kernel<<<B_ * NF_, 256, 0, stream>>>(fseq, absW + 0 * D_, fA1);
  rowdot_kernel<<<B_ * NF_, 256, 0, stream>>>(fseq, absW + 1 * D_, fA2);
  aa_pair_kernel<<<B_ * NF_ * NF_, 64, 0, stream>>>(fseq, fA1, fA2, absW + 2 * D_,
                                                    absW + 3 * D_, abs_b, aabuf);
  aa_softmax_kernel<<<B_, 256, 0, stream>>>(aabuf, aasm);

  // question-abstract scoring
  rowdot_kernel<<<B_ * NF_, 256, 0, stream>>>(fseq, scoreW + 0 * D_, fS1);
  rowdot_kernel<<<B_ * NC_, 256, 0, stream>>>(hseq, scoreW + 1 * D_, hS2);
  qa_pair_kernel<<<B_ * NC_ * NF_, 64, 0, stream>>>(fseq, hseq, fS1, hS2,
                                                    scoreW + 2 * D_, scoreW + 3 * D_,
                                                    score_b, qabuf);
  qa_softmax_kernel<<<1, B_ * NC_, 0, stream>>>(qabuf, simbuf);

  // edge weights + solver
  edge_weight_kernel<<<(B_ * NN_ * NN_ + 255) / 256, 256, 0, stream>>>(
      aa_ovl, aa_sim_in, qa_ovl, qq, aasm, simbuf,
      p_aa_ov, p_aa_sim, p_qa_ov, p_qa_sim, wedge);
  solve_edges_kernel<<<B_, 256, 0, stream>>>(wedge, fpred);

  // final loss + outputs
  loss_kernel<<<1, 256, 0, stream>>>(simbuf, gold, labels, fpred, (float*)d_out);
}
